// StructuralAttentionLayer_18064632447121
// MI455X (gfx1250) — compile-verified, matrix operand order checked by reading, not run on hardware
//
#include <hip/hip_runtime.h>
#include <hip/hip_bf16.h>
#include <stdint.h>

typedef __attribute__((ext_vector_type(16))) __bf16 v16bf;
typedef __attribute__((ext_vector_type(8)))  float  v8f;

#define IN_DIM   128
#define TOT_OUT  64
#define NHEADS   4
#define HEAD_DIM 16

__device__ __forceinline__ float lrelu02(float x) { return x > 0.f ? x : 0.2f * x; }

__device__ __forceinline__ void atomicMaxF(float* addr, float val) {
    if (val >= 0.f)
        atomicMax((int*)addr, __float_as_int(val));
    else
        atomicMin((unsigned int*)addr, __float_as_uint(val));
}

__device__ __forceinline__ void atomAddF(float* p, float v) {
    unsafeAtomicAdd(p, v);   // hardware global_atomic_add_f32
}

// ---------------------------------------------------------------------------
// Kernel 0: one-time weight fragment prep.
// Converts both weight matrices [128 x 64] (row-major f32) into WMMA
// B-fragment order (bf16), so the GEMM kernel reads each fragment with one
// 32-byte vector load instead of 16 strided scalar loads.
//
// B fragment (32x16 bf16): lane l carries column N = nt*16 + (l&15),
// lane-group lg = l>>4.
//   elems 2j,2j+1   (j<4): K = ks*32 + lg*8 + 2j (+1)
//   elems 8+2j,+1   (j<4): K = ks*32 + 16 + lg*8 + 2j (+1)
// Storage index: ((mat*4 + nt)*4 + ks)*32 + lane
// ---------------------------------------------------------------------------
__global__ void gat_prep_weights(const float* __restrict__ w0,
                                 const float* __restrict__ w1,
                                 v16bf* __restrict__ wfrag)
{
    int t = blockIdx.x * blockDim.x + threadIdx.x;
    if (t >= 2 * 4 * 4 * 32) return;
    const int lane = t & 31;
    const int ks   = (t >> 5) & 3;
    const int nt   = (t >> 7) & 3;
    const int mat  = t >> 9;
    const float* __restrict__ w = mat ? w1 : w0;

    const int col = nt * 16 + (lane & 15);
    const int lg  = lane >> 4;
    v16bf b;
#pragma unroll
    for (int j = 0; j < 4; ++j) {
        int k0 = ks * 32 + lg * 8 + 2 * j;
        b[2 * j]     = (__bf16)w[(size_t)k0       * TOT_OUT + col];
        b[2 * j + 1] = (__bf16)w[(size_t)(k0 + 1) * TOT_OUT + col];
        int k1 = ks * 32 + 16 + lg * 8 + 2 * j;
        b[8 + 2 * j]     = (__bf16)w[(size_t)k1       * TOT_OUT + col];
        b[8 + 2 * j + 1] = (__bf16)w[(size_t)(k1 + 1) * TOT_OUT + col];
    }
    wfrag[t] = b;
}

// ---------------------------------------------------------------------------
// Kernel 1: per-wave 16-node tile.  h = x@lin_w -> ws,  res = x@lin_res_w -> out
// ---------------------------------------------------------------------------
__global__ __launch_bounds__(256) void gat_node_gemm(
    const float* __restrict__ x, const v16bf* __restrict__ wfrag,
    float* __restrict__ h, float* __restrict__ out, int n_nodes)
{
    const int wave = (blockIdx.x * blockDim.x + threadIdx.x) >> 5;
    const int lane = threadIdx.x & 31;
    const int row0 = wave * 16;
    if (row0 >= n_nodes) return;

    const int r  = lane & 15;   // M row this lane carries in A
    const int kh = lane >> 4;   // K half-select
    const float* xr = x + (size_t)(row0 + r) * IN_DIM;

    // A fragments (16x32 bf16 each), K-steps ks=0..3 covering K=128.
    // elems 0..7 : K = ks*32 + kh*8 + j ;  elems 8..15 : K = ks*32 + 16 + kh*8 + j
    v16bf a[4];
#pragma unroll
    for (int ks = 0; ks < 4; ++ks) {
        const int kb = ks * 32 + kh * 8;           // 32B aligned
        float4 p0 = *(const float4*)(xr + kb);
        float4 p1 = *(const float4*)(xr + kb + 4);
        float4 p2 = *(const float4*)(xr + kb + 16);
        float4 p3 = *(const float4*)(xr + kb + 20);
        a[ks][0]  = (__bf16)p0.x; a[ks][1]  = (__bf16)p0.y;
        a[ks][2]  = (__bf16)p0.z; a[ks][3]  = (__bf16)p0.w;
        a[ks][4]  = (__bf16)p1.x; a[ks][5]  = (__bf16)p1.y;
        a[ks][6]  = (__bf16)p1.z; a[ks][7]  = (__bf16)p1.w;
        a[ks][8]  = (__bf16)p2.x; a[ks][9]  = (__bf16)p2.y;
        a[ks][10] = (__bf16)p2.z; a[ks][11] = (__bf16)p2.w;
        a[ks][12] = (__bf16)p3.x; a[ks][13] = (__bf16)p3.y;
        a[ks][14] = (__bf16)p3.z; a[ks][15] = (__bf16)p3.w;
    }

#pragma unroll
    for (int mat = 0; mat < 2; ++mat) {
        const v16bf* __restrict__ wf = wfrag + (size_t)mat * 512;
        float* __restrict__ dstp = mat ? out : h;
#pragma unroll
        for (int nt = 0; nt < 4; ++nt) {
            v8f acc = {};
#pragma unroll
            for (int ks = 0; ks < 4; ++ks) {
                v16bf b = wf[(nt * 4 + ks) * 32 + lane];   // 2x global_load_b128
                acc = __builtin_amdgcn_wmma_f32_16x16x32_bf16(
                    false, a[ks], false, b, (short)0, acc, false, false);
            }
#pragma unroll
            for (int v = 0; v < 8; ++v) {
                int m = v + kh * 8;
                dstp[(size_t)(row0 + m) * TOT_OUT + nt * 16 + (lane & 15)] = acc[v];
            }
        }
    }
}

// ---------------------------------------------------------------------------
// Kernel 2: alpha_l/alpha_r per (node, head) + init node max/sum.
// ---------------------------------------------------------------------------
__global__ void gat_alpha_init(const float* __restrict__ h,
                               const float* __restrict__ att_l,
                               const float* __restrict__ att_r,
                               float* __restrict__ alpha_l,
                               float* __restrict__ alpha_r,
                               float* __restrict__ nmax,
                               float* __restrict__ nsum, int n_nodes)
{
    int t = blockIdx.x * blockDim.x + threadIdx.x;
    if (t >= n_nodes * NHEADS) return;
    const int hd = t & 3;
    const float* hp = h + (size_t)(t >> 2) * TOT_OUT + hd * HEAD_DIM;
    float sl = 0.f, sr = 0.f;
#pragma unroll
    for (int c = 0; c < HEAD_DIM; ++c) {
        float v = hp[c];
        sl += v * att_l[hd * HEAD_DIM + c];
        sr += v * att_r[hd * HEAD_DIM + c];
    }
    alpha_l[t] = sl;
    alpha_r[t] = sr;
    nmax[t] = -INFINITY;
    nsum[t] = 0.f;
}

__global__ void gat_zero_agg(float* __restrict__ agg, int n) {
    int t = blockIdx.x * blockDim.x + threadIdx.x;
    if (t < n) agg[t] = 0.f;
}

// ---------------------------------------------------------------------------
// Kernel 3: segment max over destinations (thread = edge*head).
// ---------------------------------------------------------------------------
__global__ void gat_edge_max(const long long* __restrict__ src,
                             const long long* __restrict__ dst,
                             const float* __restrict__ ew,
                             const float* __restrict__ al,
                             const float* __restrict__ ar,
                             float* __restrict__ nmax, int n_edges)
{
    int t = blockIdx.x * blockDim.x + threadIdx.x;
    if (t >= n_edges * NHEADS) return;
    const int e = t >> 2, hd = t & 3;
    const int s = (int)src[e], d = (int)dst[e];
    float a = lrelu02(ew[e] * (al[s * NHEADS + hd] + ar[d * NHEADS + hd]));
    atomicMaxF(&nmax[d * NHEADS + hd], a);
}

// ---------------------------------------------------------------------------
// Kernel 4: segment sum of exp(alpha - max).
// ---------------------------------------------------------------------------
__global__ void gat_edge_sum(const long long* __restrict__ src,
                             const long long* __restrict__ dst,
                             const float* __restrict__ ew,
                             const float* __restrict__ al,
                             const float* __restrict__ ar,
                             const float* __restrict__ nmax,
                             float* __restrict__ nsum, int n_edges)
{
    int t = blockIdx.x * blockDim.x + threadIdx.x;
    if (t >= n_edges * NHEADS) return;
    const int e = t >> 2, hd = t & 3;
    const int s = (int)src[e], d = (int)dst[e];
    float a = lrelu02(ew[e] * (al[s * NHEADS + hd] + ar[d * NHEADS + hd]));
    float ex = __expf(a - nmax[d * NHEADS + hd]);
    atomAddF(&nsum[d * NHEADS + hd], ex);
}

// ---------------------------------------------------------------------------
// Kernel 5: message scatter. thread = (edge, head); 16 channels each.
// ---------------------------------------------------------------------------
__global__ void gat_edge_msg(const long long* __restrict__ src,
                             const long long* __restrict__ dst,
                             const float* __restrict__ ew,
                             const float* __restrict__ al,
                             const float* __restrict__ ar,
                             const float* __restrict__ nmax,
                             const float* __restrict__ nsum,
                             const float* __restrict__ h,
                             float* __restrict__ agg, int n_edges)
{
    int t = blockIdx.x * blockDim.x + threadIdx.x;
    if (t >= n_edges * NHEADS) return;
    const int e = t >> 2, hd = t & 3;
    const int s = (int)src[e], d = (int)dst[e];
    float a = lrelu02(ew[e] * (al[s * NHEADS + hd] + ar[d * NHEADS + hd]));
    float coeff = __expf(a - nmax[d * NHEADS + hd]) / nsum[d * NHEADS + hd];

    const float4* hp = (const float4*)(h + (size_t)s * TOT_OUT + hd * HEAD_DIM);
    float* ag = agg + (size_t)d * TOT_OUT + hd * HEAD_DIM;
#pragma unroll
    for (int i = 0; i < 4; ++i) {
        float4 v = hp[i];
        atomAddF(&ag[4 * i + 0], v.x * coeff);
        atomAddF(&ag[4 * i + 1], v.y * coeff);
        atomAddF(&ag[4 * i + 2], v.z * coeff);
        atomAddF(&ag[4 * i + 3], v.w * coeff);
    }
}

// ---------------------------------------------------------------------------
// Kernel 6: out = elu(agg) + res   (res already sitting in d_out)
// ---------------------------------------------------------------------------
__global__ void gat_finalize(const float* __restrict__ agg,
                             float* __restrict__ out, int n)
{
    int t = blockIdx.x * blockDim.x + threadIdx.x;
    if (t >= n) return;
    float g = agg[t];
    float e = g > 0.f ? g : (__expf(g) - 1.f);
    out[t] = e + out[t];
}

// ---------------------------------------------------------------------------
extern "C" void kernel_launch(void* const* d_in, const int* in_sizes, int n_in,
                              void* d_out, int out_size, void* d_ws, size_t ws_size,
                              hipStream_t stream)
{
    const float*     x        = (const float*)d_in[0];
    const long long* eidx     = (const long long*)d_in[1];
    const float*     ew       = (const float*)d_in[2];
    const float*     lin_w    = (const float*)d_in[3];
    const float*     att_l    = (const float*)d_in[4];
    const float*     att_r    = (const float*)d_in[5];
    const float*     lin_res  = (const float*)d_in[6];
    float*           out      = (float*)d_out;

    const int N = in_sizes[0] / IN_DIM;     // 100000
    const int E = in_sizes[2];              // 1600000
    const long long* src = eidx;
    const long long* dst = eidx + E;

    // workspace layout (floats)
    float* h       = (float*)d_ws;
    float* alpha_l = h + (size_t)N * TOT_OUT;
    float* alpha_r = alpha_l + (size_t)N * NHEADS;
    float* nmax    = alpha_r + (size_t)N * NHEADS;
    float* nsum    = nmax + (size_t)N * NHEADS;
    float* agg     = nsum + (size_t)N * NHEADS;
    // weight fragments (bf16, 32KB), 32B-aligned after agg
    v16bf* wfrag   = (v16bf*)(((uintptr_t)(agg + (size_t)N * TOT_OUT) + 31) &
                              ~(uintptr_t)31);

    const int THR = 256;
    // 0. weight fragment prep (1024 threads total)
    gat_prep_weights<<<4, THR, 0, stream>>>(lin_w, lin_res, wfrag);
    // 1. node GEMM (WMMA): 8 waves/block, 16 nodes per wave
    {
        int tiles  = (N + 15) / 16;
        int blocks = (tiles + 7) / 8;
        gat_node_gemm<<<blocks, THR, 0, stream>>>(x, wfrag, h, out, N);
    }
    // 2. alphas + init max/sum
    {
        int cnt = N * NHEADS;
        gat_alpha_init<<<(cnt + THR - 1) / THR, THR, 0, stream>>>(
            h, att_l, att_r, alpha_l, alpha_r, nmax, nsum, N);
    }
    // 3. zero agg
    {
        int cnt = N * TOT_OUT;
        gat_zero_agg<<<(cnt + THR - 1) / THR, THR, 0, stream>>>(agg, cnt);
    }
    // 4. segment max
    {
        int cnt = E * NHEADS;
        gat_edge_max<<<(cnt + THR - 1) / THR, THR, 0, stream>>>(
            src, dst, ew, alpha_l, alpha_r, nmax, E);
    }
    // 5. segment sum
    {
        int cnt = E * NHEADS;
        gat_edge_sum<<<(cnt + THR - 1) / THR, THR, 0, stream>>>(
            src, dst, ew, alpha_l, alpha_r, nmax, nsum, E);
    }
    // 6. message scatter
    {
        int cnt = E * NHEADS;
        gat_edge_msg<<<(cnt + THR - 1) / THR, THR, 0, stream>>>(
            src, dst, ew, alpha_l, alpha_r, nmax, nsum, h, agg, E);
    }
    // 7. finalize
    {
        int cnt = N * TOT_OUT;
        gat_finalize<<<(cnt + THR - 1) / THR, THR, 0, stream>>>(agg, out, cnt);
    }
}
